// OGFormer_4861902979532
// MI455X (gfx1250) — matrix operand order repaired and checked by reading, not compiled, not run on hardware
//
#include <hip/hip_runtime.h>
#include <cstdint>

#define NN_  6144
#define FIN_ 256
#define DIM_ 64
#define RT_  (NN_/16)   // 384 row tiles

typedef _Float16 h16;
typedef __attribute__((ext_vector_type(16))) _Float16 v16h;
typedef __attribute__((ext_vector_type(8)))  _Float16 v8h;
typedef __attribute__((ext_vector_type(8)))  float    v8f;

__device__ __forceinline__ v8f wmma16(v16h a, v16h b, v8f c) {
  return __builtin_amdgcn_wmma_f32_16x16x32_f16(false, a, false, b, (short)0, c, false, false);
}

__device__ __forceinline__ v16h cat8(v8h lo, v8h hi) {
  v16h r;
#pragma unroll
  for (int i = 0; i < 8; ++i) { r[i] = lo[i]; r[i + 8] = hi[i]; }
  return r;
}

// ---- operand builders (ISA 7.12.2 layouts, wave32) -------------------------
// A (16x32 f16): lane m=lane&15; lanes 0-15 K={0..7,16..23}, lanes 16-31 K={8..15,24..31}
__device__ __forceinline__ v16h load_a_h(const h16* __restrict__ src, int ld,
                                         int rowBase, int kBase, int lane) {
  const int m  = lane & 15;
  const int k0 = kBase + ((lane & 16) ? 8 : 0);
  const h16* p = src + (size_t)(rowBase + m) * (size_t)ld + k0;
  return cat8(*(const v8h*)p, *(const v8h*)(p + 16));
}
// A from f32 source (convert on the fly; contiguous 128-bit loads)
__device__ __forceinline__ v16h load_a_f(const float* __restrict__ src, int ld,
                                         int rowBase, int kBase, int lane) {
  const int m  = lane & 15;
  const int k0 = kBase + ((lane & 16) ? 8 : 0);
  const float* p = src + (size_t)(rowBase + m) * (size_t)ld + k0;
  float4 x = *(const float4*)(p);
  float4 y = *(const float4*)(p + 4);
  float4 z = *(const float4*)(p + 16);
  float4 w = *(const float4*)(p + 20);
  v16h a;
  a[0]=(h16)x.x;  a[1]=(h16)x.y;  a[2]=(h16)x.z;  a[3]=(h16)x.w;
  a[4]=(h16)y.x;  a[5]=(h16)y.y;  a[6]=(h16)y.z;  a[7]=(h16)y.w;
  a[8]=(h16)z.x;  a[9]=(h16)z.y;  a[10]=(h16)z.z; a[11]=(h16)z.w;
  a[12]=(h16)w.x; a[13]=(h16)w.y; a[14]=(h16)w.z; a[15]=(h16)w.w;
  return a;
}
// B (32x16 f16) from ROWS of an f16 matrix: B[k][n] = src[rowOff+n][kBase+k]
__device__ __forceinline__ v16h load_b_h(const h16* __restrict__ src, int ld,
                                         int rowOff, int kBase, int lane) {
  const int n  = lane & 15;
  const int k0 = kBase + ((lane & 16) ? 8 : 0);
  const h16* p = src + (size_t)(rowOff + n) * (size_t)ld + k0;
  return cat8(*(const v8h*)p, *(const v8h*)(p + 16));
}
// B from a small row-major f32 weight (Kdim x 64): strided, used only in proj
__device__ __forceinline__ v16h load_b_w(const float* __restrict__ src, int ld,
                                         int kBase, int colBase, int lane) {
  const int n  = colBase + (lane & 15);
  const int k0 = kBase + ((lane & 16) ? 8 : 0);
  v16h b;
#pragma unroll
  for (int i = 0; i < 16; ++i) b[i] = (h16)src[(size_t)(k0 + (i < 8 ? i : i + 8)) * (size_t)ld + n];
  return b;
}

// reductions within each 16-lane half (matches C-matrix row striping on wave32)
__device__ __forceinline__ float hmax16(float v) {
  v = fmaxf(v, __shfl_xor(v, 1, 32));
  v = fmaxf(v, __shfl_xor(v, 2, 32));
  v = fmaxf(v, __shfl_xor(v, 4, 32));
  v = fmaxf(v, __shfl_xor(v, 8, 32));
  return v;
}
__device__ __forceinline__ float hsum16(float v) {
  v += __shfl_xor(v, 1, 32);
  v += __shfl_xor(v, 2, 32);
  v += __shfl_xor(v, 4, 32);
  v += __shfl_xor(v, 8, 32);
  return v;
}

// ---- CDNA5 async global -> LDS (ASYNCcnt) ----------------------------------
__device__ __forceinline__ void async_ld_b128(uint32_t lds_off, const void* g) {
  asm volatile("global_load_async_to_lds_b128 %0, %1, off"
               :: "v"(lds_off), "v"((unsigned long long)(uintptr_t)g) : "memory");
}
__device__ __forceinline__ void wait_async0() {
  asm volatile("s_wait_asynccnt 0" ::: "memory");
}

// ---------------------------------------------------------------------------
__global__ void zero_kernel(float* __restrict__ p, size_t n) {
  size_t i = (size_t)blockIdx.x * blockDim.x + threadIdx.x;
  if (i < n) p[i] = 0.0f;
}

// ---------------------------------------------------------------------------
// Projections: Q = X@Wq (f32 + f16 copies), K = X@Wk (f16), AXt = (X@Wa)^T (f16).
template <bool A_HALF>
__global__ void proj_kernel(const float* __restrict__ Xf, const h16* __restrict__ Xh,
                            int Kdim,
                            const float* __restrict__ Wq, const float* __restrict__ Wk,
                            const float* __restrict__ Wa,
                            float* __restrict__ Qf, h16* __restrict__ Qh,
                            h16* __restrict__ Kh, h16* __restrict__ AXt) {
  const int lane = threadIdx.x & 31;
  const int wave = threadIdx.x >> 5;
  const int rowBase = (blockIdx.x * 8 + wave) * 16;
  const float* W[3] = {Wq, Wk, Wa};
  const int half8 = (lane & 16) ? 8 : 0;
  const int col0  = lane & 15;
#pragma unroll
  for (int w = 0; w < 3; ++w) {
    v8f acc[4] = {v8f{}, v8f{}, v8f{}, v8f{}};
    for (int k = 0; k < Kdim; k += 32) {
      v16h a = A_HALF ? load_a_h(Xh, Kdim, rowBase, k, lane)
                      : load_a_f(Xf, Kdim, rowBase, k, lane);
#pragma unroll
      for (int c = 0; c < 4; ++c)
        acc[c] = wmma16(a, load_b_w(W[w], DIM_, k, c * 16, lane), acc[c]);
    }
#pragma unroll
    for (int c = 0; c < 4; ++c)
#pragma unroll
      for (int r = 0; r < 8; ++r) {
        const int row = rowBase + r + half8;
        const int col = c * 16 + col0;
        const float v = acc[c][r];
        if (w == 0) { Qf[(size_t)row * DIM_ + col] = v; Qh[(size_t)row * DIM_ + col] = (h16)v; }
        else if (w == 1) { Kh[(size_t)row * DIM_ + col] = (h16)v; }
        else { AXt[(size_t)col * NN_ + row] = (h16)v; }   // transposed for B-operand reads
      }
  }
}

// ---------------------------------------------------------------------------
// Softmax stats of QK^T/8, two per-lane passes (recompute instead of online):
// pass A: per-lane running max, one cross-lane reduce at the end;
// pass B: per-lane sum of exp(c*0.125 - m), one cross-lane reduce at the end.
// K rows staged 128 at a time into LDS via CDNA5 async copies, shared by 8 waves.
__global__ void attn_stats_kernel(const h16* __restrict__ Qh, const h16* __restrict__ Kh,
                                  float* __restrict__ rowmax, float* __restrict__ rowsum) {
  __shared__ h16 kbuf[128 * DIM_];                 // 16 KB
  const int lane = threadIdx.x & 31;
  const int wave = threadIdx.x >> 5;
  const int rowBase = (blockIdx.x * 8 + wave) * 16;
  const uint32_t lbase = (uint32_t)(uintptr_t)&kbuf[0];

  v16h a0 = load_a_h(Qh, DIM_, rowBase, 0, lane);
  v16h a1 = load_a_h(Qh, DIM_, rowBase, 32, lane);

  float mloc[8];
#pragma unroll
  for (int r = 0; r < 8; ++r) mloc[r] = -__builtin_inff();

  // ---- pass A: raw per-lane max ----
  for (int jc = 0; jc < NN_; jc += 128) {
    const char* gsrc = (const char*)(Kh + (size_t)jc * DIM_);
#pragma unroll
    for (int i = 0; i < 4; ++i) {
      const uint32_t off = (uint32_t)threadIdx.x * 64u + i * 16u;
      async_ld_b128(lbase + off, gsrc + off);
    }
    wait_async0();
    __syncthreads();
#pragma unroll 1
    for (int j8 = 0; j8 < 8; ++j8) {
      const h16* kb = &kbuf[j8 * 16 * DIM_];
      v8f c = v8f{};
      c = wmma16(a0, load_b_h(kb, DIM_, 0, 0, lane), c);
      c = wmma16(a1, load_b_h(kb, DIM_, 0, 32, lane), c);
#pragma unroll
      for (int r = 0; r < 8; ++r) mloc[r] = fmaxf(mloc[r], c[r]);
    }
    __syncthreads();
  }
  float m[8];
#pragma unroll
  for (int r = 0; r < 8; ++r) m[r] = hmax16(mloc[r]) * 0.125f;  // broadcast to all 16 lanes

  // ---- pass B: per-lane sum of exp ----
  float sl[8];
#pragma unroll
  for (int r = 0; r < 8; ++r) sl[r] = 0.0f;
  for (int jc = 0; jc < NN_; jc += 128) {
    const char* gsrc = (const char*)(Kh + (size_t)jc * DIM_);
#pragma unroll
    for (int i = 0; i < 4; ++i) {
      const uint32_t off = (uint32_t)threadIdx.x * 64u + i * 16u;
      async_ld_b128(lbase + off, gsrc + off);
    }
    wait_async0();
    __syncthreads();
#pragma unroll 1
    for (int j8 = 0; j8 < 8; ++j8) {
      const h16* kb = &kbuf[j8 * 16 * DIM_];
      v8f c = v8f{};
      c = wmma16(a0, load_b_h(kb, DIM_, 0, 0, lane), c);
      c = wmma16(a1, load_b_h(kb, DIM_, 0, 32, lane), c);
#pragma unroll
      for (int r = 0; r < 8; ++r) sl[r] += __expf(__fmaf_rn(c[r], 0.125f, -m[r]));
    }
    __syncthreads();
  }

  if ((lane & 15) == 0) {
    const int half8 = (lane & 16) ? 8 : 0;
#pragma unroll
    for (int r = 0; r < 8; ++r) {
      rowmax[rowBase + r + half8] = m[r];
      rowsum[rowBase + r + half8] = hsum16(sl[r]);
    }
  }
  else {
#pragma unroll
    for (int r = 0; r < 8; ++r) (void)hsum16(sl[r]);
  }
}

// ---------------------------------------------------------------------------
// R = alpha*adjPrev + (1-alpha)*softmax(QK^T/8); accumulate row degrees.
// grid = (RT_, 8), block = 256; wave handles (rowTile, 96-column chunk).
__global__ void attn_write_kernel(const h16* __restrict__ Qh, const h16* __restrict__ Kh,
                                  const float* __restrict__ adjPrev,
                                  const float* __restrict__ rowmax,
                                  const float* __restrict__ rowsum,
                                  float alpha, float* __restrict__ R,
                                  float* __restrict__ deg) {
  const int lane = threadIdx.x & 31;
  const int wave = threadIdx.x >> 5;
  const int rowBase = blockIdx.x * 16;
  const int jStart  = (blockIdx.y * 8 + wave) * 96;   // 6 j-tiles of 16
  const int half8 = (lane & 16) ? 8 : 0;
  const int col0  = lane & 15;
  v16h a0 = load_a_h(Qh, DIM_, rowBase, 0, lane);
  v16h a1 = load_a_h(Qh, DIM_, rowBase, 32, lane);
  float rm[8], inv[8], part[8];
#pragma unroll
  for (int r = 0; r < 8; ++r) {
    rm[r]   = rowmax[rowBase + r + half8];
    inv[r]  = 1.0f / rowsum[rowBase + r + half8];
    part[r] = 0.0f;
  }
  const float beta = 1.0f - alpha;
#pragma unroll 1
  for (int t = 0; t < 6; ++t) {
    const int jBase = jStart + t * 16;
    v8f c = v8f{};
    c = wmma16(a0, load_b_h(Kh, DIM_, jBase, 0, lane), c);
    c = wmma16(a1, load_b_h(Kh, DIM_, jBase, 32, lane), c);
#pragma unroll
    for (int r = 0; r < 8; ++r) {
      const size_t idx = (size_t)(rowBase + r + half8) * NN_ + jBase + col0;
      __builtin_prefetch(&adjPrev[idx + 16], 0, 3);   // next tile, same row stream
      float attn = __expf(__fmaf_rn(c[r], 0.125f, -rm[r])) * inv[r];
      float rv   = alpha * adjPrev[idx] + beta * attn;
      R[idx]     = rv;
      part[r]   += rv;
    }
  }
#pragma unroll
  for (int r = 0; r < 8; ++r) {
    float p = hsum16(part[r]);
    if ((lane & 15) == 0) atomicAdd(&deg[rowBase + r + half8], p);
  }
}

// ---------------------------------------------------------------------------
// Hacc += R[rowTile, kSlice] @ AX  (B read from transposed f16 AXt).
__global__ void agg_partial_kernel(const float* __restrict__ R, const h16* __restrict__ AXt,
                                   float* __restrict__ Hacc) {
  const int lane = threadIdx.x & 31;
  const int wave = threadIdx.x >> 5;
  const int rowBase = (blockIdx.x * 8 + wave) * 16;
  const int kt0 = blockIdx.y * 24;       // 192 k-tiles of 32, split 8 ways
  v8f acc[4] = {v8f{}, v8f{}, v8f{}, v8f{}};
#pragma unroll 1
  for (int kt = kt0; kt < kt0 + 24; ++kt) {
    const int k = kt * 32;
    v16h a = load_a_f(R, NN_, rowBase, k, lane);
#pragma unroll
    for (int c = 0; c < 4; ++c)
      acc[c] = wmma16(a, load_b_h(AXt, NN_, c * 16, k, lane), acc[c]);
  }
  const int half8 = (lane & 16) ? 8 : 0;
  const int col0  = lane & 15;
#pragma unroll
  for (int c = 0; c < 4; ++c)
#pragma unroll
    for (int r = 0; r < 8; ++r)
      atomicAdd(&Hacc[(size_t)(rowBase + r + half8) * DIM_ + c * 16 + col0], acc[c][r]);
}

// ---------------------------------------------------------------------------
__global__ void finalize_kernel(const float* __restrict__ Hacc, const float* __restrict__ deg,
                                const float* __restrict__ Qadd, int doRelu,
                                float* __restrict__ H, h16* __restrict__ Hh) {
  const int i = blockIdx.x * blockDim.x + threadIdx.x;   // exactly N*64 threads
  const int row = i >> 6;
  float d = fmaxf(deg[row], 1e-12f);
  float v = Hacc[i] / d;
  if (doRelu) v = fmaxf(v, 0.0f);
  v += Qadd[i];
  H[i] = v;
  if (Hh) Hh[i] = (h16)v;
}

// ---------------------------------------------------------------------------
__global__ void logsoftmax_kernel(const float* __restrict__ H, float* __restrict__ Out) {
  const int lane = threadIdx.x & 31;
  const int wave = threadIdx.x >> 5;
  const int row  = blockIdx.x * 8 + wave;
  float x0 = H[(size_t)row * DIM_ + lane];
  float x1 = H[(size_t)row * DIM_ + 32 + lane];
  float mx = fmaxf(x0, x1);
#pragma unroll
  for (int m = 1; m <= 16; m <<= 1) mx = fmaxf(mx, __shfl_xor(mx, m, 32));
  float e = __expf(x0 - mx) + __expf(x1 - mx);
#pragma unroll
  for (int m = 1; m <= 16; m <<= 1) e += __shfl_xor(e, m, 32);
  float lse = mx + __logf(e);
  Out[(size_t)row * DIM_ + lane]      = x0 - lse;
  Out[(size_t)row * DIM_ + 32 + lane] = x1 - lse;
}

// ---------------------------------------------------------------------------
extern "C" void kernel_launch(void* const* d_in, const int* in_sizes, int n_in,
                              void* d_out, int out_size, void* d_ws, size_t ws_size,
                              hipStream_t stream) {
  const float* adj  = (const float*)d_in[0];
  const float* feat = (const float*)d_in[1];
  const float* Wq1  = (const float*)d_in[2];
  const float* Wk1  = (const float*)d_in[3];
  const float* Wa1  = (const float*)d_in[4];
  const float* Wq2  = (const float*)d_in[5];
  const float* Wk2  = (const float*)d_in[6];
  const float* Wa2  = (const float*)d_in[7];

  const size_t ND  = (size_t)NN_ * DIM_;
  const size_t NNs = (size_t)NN_ * NN_;
  float* out       = (float*)d_out;
  float* out_logsm = out;                  // log_softmax(h2)
  float* out_hq1   = out + ND;             // hq1
  float* out_R1    = out + 2 * ND;         // R1
  float* out_hq2   = out + 2 * ND + NNs;   // hq2
  float* out_R2    = out + 3 * ND + NNs;   // R2

  // workspace: 7*ND halves (f16) then f32 region
  h16* Qh1  = (h16*)d_ws;
  h16* Kh1  = Qh1 + ND;
  h16* AXt1 = Kh1 + ND;
  h16* Qh2  = AXt1 + ND;
  h16* Kh2  = Qh2 + ND;
  h16* AXt2 = Kh2 + ND;
  h16* h1h  = AXt2 + ND;
  float* fb   = (float*)(h1h + ND);
  float* h1   = fb;
  float* h2   = h1 + ND;
  float* Hacc1 = h2 + ND;
  float* Hacc2 = Hacc1 + ND;
  float* rm1  = Hacc2 + ND;
  float* rs1  = rm1 + NN_;
  float* deg1 = rm1 + 2 * NN_;
  float* rm2  = rm1 + 3 * NN_;
  float* rs2  = rm1 + 4 * NN_;
  float* deg2 = rm1 + 5 * NN_;

  // zero Hacc1, Hacc2 and deg region (contiguous)
  const size_t zn = 2 * ND + 6 * (size_t)NN_;
  zero_kernel<<<(unsigned)((zn + 255) / 256), 256, 0, stream>>>(Hacc1, zn);

  // ---- layer 1 ----
  proj_kernel<false><<<RT_ / 8, 256, 0, stream>>>(feat, nullptr, FIN_, Wq1, Wk1, Wa1,
                                                  out_hq1, Qh1, Kh1, AXt1);
  attn_stats_kernel<<<RT_ / 8, 256, 0, stream>>>(Qh1, Kh1, rm1, rs1);
  attn_write_kernel<<<dim3(RT_, 8), 256, 0, stream>>>(Qh1, Kh1, adj, rm1, rs1,
                                                      0.8f, out_R1, deg1);
  agg_partial_kernel<<<dim3(RT_ / 8, 8), 256, 0, stream>>>(out_R1, AXt1, Hacc1);
  finalize_kernel<<<(unsigned)(ND / 256), 256, 0, stream>>>(Hacc1, deg1, out_hq1, 1, h1, h1h);

  // ---- layer 2 ----
  proj_kernel<true><<<RT_ / 8, 256, 0, stream>>>(nullptr, h1h, DIM_, Wq2, Wk2, Wa2,
                                                 out_hq2, Qh2, Kh2, AXt2);
  attn_stats_kernel<<<RT_ / 8, 256, 0, stream>>>(Qh2, Kh2, rm2, rs2);
  attn_write_kernel<<<dim3(RT_, 8), 256, 0, stream>>>(Qh2, Kh2, out_R1, rm2, rs2,
                                                      0.9f, out_R2, deg2);
  agg_partial_kernel<<<dim3(RT_ / 8, 8), 256, 0, stream>>>(out_R2, AXt2, Hacc2);
  finalize_kernel<<<(unsigned)(ND / 256), 256, 0, stream>>>(Hacc2, deg2, out_hq2, 0, h2, nullptr);

  logsoftmax_kernel<<<NN_ / 8, 256, 0, stream>>>(h2, out_logsm);
}
